// VSSBlock_17643725652508
// MI455X (gfx1250) — matrix operand order repaired and checked by reading, not compiled, not run on hardware
//
#include <hip/hip_runtime.h>
#include <hip/hip_bf16.h>

// ---------------------------------------------------------------------------
// VSSBlock (VMamba) forward for MI455X (gfx1250, wave32, WMMA).
//   * in_proj / x_proj / out_proj on v_wmma_f32_16x16x32_f16, f32 accum.
//   * 8-wave workgroups; weight tile staged once into LDS (ds_load_b128 feeds
//     the B fragments), A fragments stream from global as b128.
//   * dt_proj + softplus fused into the selective scan; cross-merge done with
//     global_atomic_add_f32 through an incremental direction permutation.
// ---------------------------------------------------------------------------

typedef __attribute__((ext_vector_type(16))) _Float16 v16h;
typedef __attribute__((ext_vector_type(8)))  _Float16 v8h;
typedef __attribute__((ext_vector_type(8)))  float    v8f;

constexpr int cB   = 4;
constexpr int cC   = 128;
constexpr int cH   = 48;
constexpr int cW   = 48;
constexpr int cL   = cH * cW;      // 2304
constexpr int cDI  = 256;
constexpr int cNS  = 16;
constexpr int cR   = 8;
constexpr int cK   = 4;
constexpr int cNPIX = cB * cL;     // 9216
constexpr int cLB  = cL / 128;     // 18 (128-row chunks per batch image)

__device__ __forceinline__ float silu_f(float x) {
    // x * sigmoid(x) with fast reciprocal (v_rcp_f32), no IEEE divide.
    return x * __builtin_amdgcn_rcpf(1.f + __expf(-x));
}

__device__ __forceinline__ v16h pack16(v8h lo, v8h hi) {
    v16h a;
#pragma unroll
    for (int i = 0; i < 8; ++i) { a[i] = lo[i]; a[i + 8] = hi[i]; }
    return a;
}

// ----------------------------- small helpers ------------------------------

__global__ void cvt16(const float* __restrict__ s, _Float16* __restrict__ d, int n) {
    int i = blockIdx.x * blockDim.x + threadIdx.x;
    if (i < n) d[i] = (_Float16)s[i];
}

// LayerNorm over C (eps 1e-6) -> f16 activations (NPIX, C)
__global__ void ln_in(const float* __restrict__ x, const float* __restrict__ w,
                      const float* __restrict__ bv, _Float16* __restrict__ xn) {
    int p = blockIdx.x * blockDim.x + threadIdx.x;
    if (p >= cNPIX) return;
    int b = p / cL, hw = p % cL;
    const float* xp = x + (size_t)b * cC * cL + hw;
    float s = 0.f, s2 = 0.f;
    for (int c = 0; c < cC; ++c) { float v = xp[(size_t)c * cL]; s += v; s2 += v * v; }
    float mu = s * (1.f / cC);
    float var = s2 * (1.f / cC) - mu * mu;
    float rinv = rsqrtf(var + 1e-6f);
    _Float16* op = xn + (size_t)p * cC;
    for (int c = 0; c < cC; ++c) {
        float v = (xp[(size_t)c * cL] - mu) * rinv * w[c] + bv[c];
        op[c] = (_Float16)v;
    }
}

// ------------------------- GEMM 1: LN(x) @ in_proj^T ----------------------
// 8 waves/block: wave w computes rows [bx*128 + 16w, +16), all share one
// 16-column weight tile staged in LDS. The x1/z split is uniform per block
// (n0 is a multiple of 16), so the epilogue branch is scalar.
__global__ __launch_bounds__(256) void gemm_inproj(
    const _Float16* __restrict__ xn,   // (NPIX, C)
    const _Float16* __restrict__ w1,   // (2*DI, C)
    float* __restrict__ x1,            // (B, DI, L)  conv input (NCHW)
    float* __restrict__ z)             // (B, L, DI)  silu gate
{
    __shared__ _Float16 sw[16 * cC];               // 4 KB weight tile
    const int tid  = threadIdx.x;
    const int lane = tid & 31;
    const int wv   = tid >> 5;
    const int lo = lane & 15, hi = lane >> 4;
    const int n0 = blockIdx.y * 16;

    // cooperative weight-tile load: rows n0..n0+15 are contiguous in memory
    *(v8h*)(&sw[tid * 8]) = *(const v8h*)(w1 + (size_t)n0 * cC + tid * 8);
    __syncthreads();

    const int m0 = blockIdx.x * 128 + wv * 16;
    v8f acc = {};
#pragma unroll
    for (int kt = 0; kt < 4; ++kt) {
        const _Float16* ap = xn + (size_t)(m0 + lo) * cC + kt * 32 + 8 * hi;
        v16h a = pack16(*(const v8h*)(ap), *(const v8h*)(ap + 16));
        const _Float16* bp = &sw[lo * cC + kt * 32 + 8 * hi];
        v16h bf = pack16(*(const v8h*)(bp), *(const v8h*)(bp + 16));
        acc = __builtin_amdgcn_wmma_f32_16x16x32_f16(false, a, false, bf,
                                                     (short)0, acc, false, false);
    }
    // scalar (uniform) batch / pixel decomposition: 2304 == 18*128
    const int b   = blockIdx.x / cLB;
    const int hwb = (blockIdx.x % cLB) * 128 + wv * 16;
    const int n   = n0 + lo;
    if (n0 < cDI) {                                 // uniform: x1 half
#pragma unroll
        for (int r = 0; r < 8; ++r) {
            int hw = hwb + r + 8 * hi;
            x1[((size_t)b * cDI + n) * cL + hw] = acc[r];
        }
    } else {                                        // uniform: gate half
#pragma unroll
        for (int r = 0; r < 8; ++r) {
            int hw = hwb + r + 8 * hi;
            z[((size_t)b * cL + hw) * cDI + (n - cDI)] = silu_f(acc[r]);
        }
    }
}

// ---------------- depthwise 3x3 conv + bias + SiLU -> f16 (B,L,DI) --------
__global__ void dwconv(const float* __restrict__ x1, const float* __restrict__ cw,
                       const float* __restrict__ cb, _Float16* __restrict__ xc16) {
    int hw = blockIdx.x * blockDim.x + threadIdx.x;
    int bd = blockIdx.y;                  // b*DI + d
    if (hw >= cL) return;
    int b = bd >> 8, d = bd & 255;
    int h = hw / cW, w = hw % cW;
    const float* src = x1 + (size_t)bd * cL;
    const float* wk  = cw + d * 9;
    float acc = cb[d];
#pragma unroll
    for (int dy = -1; dy <= 1; ++dy)
#pragma unroll
        for (int dx = -1; dx <= 1; ++dx) {
            int hh = h + dy, ww = w + dx;
            if (hh >= 0 && hh < cH && ww >= 0 && ww < cW)
                acc += src[hh * cW + ww] * wk[(dy + 1) * 3 + (dx + 1)];
        }
    xc16[((size_t)b * cL + hw) * cDI + d] = (_Float16)silu_f(acc);
}

// ----------------- GEMM 2: x_proj per direction (gathered A) --------------
// xdbl[b,k,c',l] = sum_d xs[b,k,d,l] * x_proj_w[k][c'][d],  c' in [0,40)
__global__ __launch_bounds__(256) void gemm_xproj(
    const _Float16* __restrict__ xc16, // (B, L, DI)
    const _Float16* __restrict__ w2,   // (K, 40, DI)
    float* __restrict__ xdbl)          // (B, K, 40, L)
{
    __shared__ _Float16 sw[16 * cDI];              // 8 KB weight tile
    const int tid  = threadIdx.x;
    const int lane = tid & 31;
    const int wv   = tid >> 5;
    const int lo = lane & 15, hi = lane >> 4;
    const int n0 = blockIdx.y * 16;                // 0,16,32 (valid n < 40)
    const int b  = blockIdx.z >> 2;
    const int kd = blockIdx.z & 3;

    // cooperative weight-tile load with zero-fill past row 39
#pragma unroll
    for (int c2 = 0; c2 < 2; ++c2) {
        int off = tid * 16 + c2 * 8;
        int nr  = n0 + (off >> 8);                 // off/cDI
        v8h v;
        if (nr < 40) v = *(const v8h*)(w2 + ((size_t)kd * 40 + nr) * cDI + (off & 255));
        else {
#pragma unroll
            for (int i = 0; i < 8; ++i) v[i] = (_Float16)0.f;
        }
        *(v8h*)(&sw[off]) = v;
    }
    __syncthreads();

    const int m0 = blockIdx.x * 128 + wv * 16;     // sequence-position tile
    // direction permutation for this wave's 16 A-rows (one div per thread)
    int l = m0 + lo;
    int ll = (kd & 2) ? (cL - 1 - l) : l;
    int pix = (kd & 1) ? ((ll % cH) * cW + ll / cH) : ll;
    const _Float16* abase = xc16 + ((size_t)b * cL + pix) * cDI;

    v8f acc = {};
#pragma unroll
    for (int kt = 0; kt < 8; ++kt) {
        const _Float16* ap = abase + kt * 32 + 8 * hi;
        v16h a = pack16(*(const v8h*)(ap), *(const v8h*)(ap + 16));
        const _Float16* bp = &sw[lo * cDI + kt * 32 + 8 * hi];
        v16h bf = pack16(*(const v8h*)(bp), *(const v8h*)(bp + 16));
        acc = __builtin_amdgcn_wmma_f32_16x16x32_f16(false, a, false, bf,
                                                     (short)0, acc, false, false);
    }
    const int n = n0 + lo;
    if (n < 40) {
#pragma unroll
        for (int r = 0; r < 8; ++r) {
            int lr = m0 + r + 8 * hi;
            xdbl[(((size_t)b * cK + kd) * 40 + n) * cL + lr] = acc[r];
        }
    }
}

// ----------------------------- selective scan -----------------------------
// One wave32 per (b, direction, 32 d-channels). dt_proj+softplus fused.
// B/C/dts are per-(b,k,l) scalars: fetched once per wave, broadcast by shfl.
// Cross-merge via global_atomic_add_f32; permutation updated incrementally.
__global__ __launch_bounds__(32) void sel_scan(
    const float* __restrict__ xdbl,    // (B,K,40,L): rows 0-7 dts, 8-23 B, 24-39 C
    const _Float16* __restrict__ xc16, // (B,L,DI)   u values
    const float* __restrict__ dtw,     // (K,DI,R)
    const float* __restrict__ dtb,     // (K,DI)
    const float* __restrict__ alog,    // (K*DI, NS)
    const float* __restrict__ Ds,      // (K*DI)
    float* __restrict__ ymg)           // (B,L,DI)
{
    const int lane = threadIdx.x;
    const int wid  = blockIdx.x;            // 0..127 = b*32 + kd*8 + dgroup
    const int b  = wid >> 5;
    const int kd = (wid >> 3) & 3;
    const int d  = ((wid & 7) << 5) | lane;
    const int ch = kd * cDI + d;

    float wdt[cR];
#pragma unroll
    for (int r = 0; r < cR; ++r) wdt[r] = dtw[(size_t)ch * cR + r];
    const float bdt = dtb[ch];
    float Arow[cNS];
#pragma unroll
    for (int n2 = 0; n2 < cNS; ++n2) Arow[n2] = -__expf(alog[(size_t)ch * cNS + n2]);
    const float dsc = Ds[ch];
    float hst[cNS];
#pragma unroll
    for (int n2 = 0; n2 < cNS; ++n2) hst[n2] = 0.f;

    const float* xb    = xdbl + ((size_t)b * cK + kd) * 40 * cL;
    const float* bcrow = xb + (size_t)(8 + lane) * cL;   // lanes 0-15: B[n], 16-31: C[n]
    const float* dtrow = xb + (size_t)(lane & 7) * cL;   // dts[r]
    const _Float16* ub = xc16 + (size_t)b * cL * cDI;
    float* yb = ymg + (size_t)b * cL * cDI;

    // incremental direction permutation: no div/mod in the loop
    int pix = (kd & 2) ? (cL - 1) : 0;
    const int step = (kd & 1) ? cW : 1;
    const int dp   = (kd & 2) ? -step : step;
    const bool wrap = (kd & 1);

    for (int l = 0; l < cL; ++l) {
        __builtin_prefetch(bcrow + l + 16, 0, 0);        // global_prefetch_b8
        float bc  = bcrow[l];
        float dtv = dtrow[l];
        float pre = bdt;
#pragma unroll
        for (int r = 0; r < cR; ++r) pre += wdt[r] * __shfl(dtv, r, 32);
        float delta = (pre > 20.f) ? pre : __logf(1.f + __expf(pre));
        float u  = (float)ub[(size_t)pix * cDI + d];
        float du = delta * u;
        float y = 0.f;
#pragma unroll
        for (int n2 = 0; n2 < cNS; ++n2) {
            float Bv = __shfl(bc, n2, 32);
            float Cv = __shfl(bc, 16 + n2, 32);
            hst[n2] = __expf(delta * Arow[n2]) * hst[n2] + du * Bv;
            y += hst[n2] * Cv;
        }
        atomicAdd(&yb[(size_t)pix * cDI + d], y + u * dsc);
        pix += dp;
        if (wrap) {
            if (pix >= cL) pix -= (cL - 1);
            else if (pix < 0) pix += (cL - 1);
        }
    }
}

// ------------- out-LayerNorm (eps 1e-5) * w + b, gate by z -> f16 ----------
__global__ void ln_out(const float* __restrict__ ymg, const float* __restrict__ w,
                       const float* __restrict__ bv, const float* __restrict__ z,
                       _Float16* __restrict__ yln) {
    int p = blockIdx.x * blockDim.x + threadIdx.x;
    if (p >= cNPIX) return;
    const float* yp = ymg + (size_t)p * cDI;
    float s = 0.f, s2 = 0.f;
    for (int d = 0; d < cDI; ++d) { float v = yp[d]; s += v; s2 += v * v; }
    float mu = s * (1.f / cDI);
    float var = s2 * (1.f / cDI) - mu * mu;
    float rinv = rsqrtf(var + 1e-5f);
    const float* zp = z + (size_t)p * cDI;
    _Float16* op = yln + (size_t)p * cDI;
    for (int d = 0; d < cDI; ++d) {
        float v = ((yp[d] - mu) * rinv * w[d] + bv[d]) * zp[d];
        op[d] = (_Float16)v;
    }
}

// -------------------- GEMM 3: out_proj + residual -------------------------
__global__ __launch_bounds__(256) void gemm_outproj(
    const _Float16* __restrict__ yln,  // (NPIX, DI)
    const _Float16* __restrict__ w3,   // (C, DI)
    const float* __restrict__ xin,     // (B, C, H, W) residual
    float* __restrict__ out)           // (B, C, H, W)
{
    __shared__ _Float16 sw[16 * cDI];              // 8 KB weight tile
    const int tid  = threadIdx.x;
    const int lane = tid & 31;
    const int wv   = tid >> 5;
    const int lo = lane & 15, hi = lane >> 4;
    const int n0 = blockIdx.y * 16;

    // cooperative weight-tile load: rows contiguous -> flat b128 copies
#pragma unroll
    for (int c2 = 0; c2 < 2; ++c2) {
        int off = tid * 16 + c2 * 8;
        *(v8h*)(&sw[off]) = *(const v8h*)(w3 + (size_t)n0 * cDI + off);
    }
    __syncthreads();

    const int m0 = blockIdx.x * 128 + wv * 16;
    v8f acc = {};
#pragma unroll
    for (int kt = 0; kt < 8; ++kt) {
        const _Float16* ap = yln + (size_t)(m0 + lo) * cDI + kt * 32 + 8 * hi;
        v16h a = pack16(*(const v8h*)(ap), *(const v8h*)(ap + 16));
        const _Float16* bp = &sw[lo * cDI + kt * 32 + 8 * hi];
        v16h bf = pack16(*(const v8h*)(bp), *(const v8h*)(bp + 16));
        acc = __builtin_amdgcn_wmma_f32_16x16x32_f16(false, a, false, bf,
                                                     (short)0, acc, false, false);
    }
    const int b   = blockIdx.x / cLB;
    const int hwb = (blockIdx.x % cLB) * 128 + wv * 16;
    const int n   = n0 + lo;
#pragma unroll
    for (int r = 0; r < 8; ++r) {
        int hw = hwb + r + 8 * hi;
        size_t idx = ((size_t)b * cC + n) * cL + hw;
        out[idx] = xin[idx] + acc[r];
    }
}

// ---------------------------------------------------------------------------

extern "C" void kernel_launch(void* const* d_in, const int* in_sizes, int n_in,
                              void* d_out, int out_size, void* d_ws, size_t ws_size,
                              hipStream_t stream) {
    (void)in_sizes; (void)n_in; (void)out_size; (void)ws_size;
    const float* x     = (const float*)d_in[0];
    const float* lnw   = (const float*)d_in[1];
    const float* lnb   = (const float*)d_in[2];
    const float* w_in  = (const float*)d_in[3];
    const float* cw    = (const float*)d_in[4];
    const float* cb    = (const float*)d_in[5];
    const float* w_xp  = (const float*)d_in[6];
    const float* w_dt  = (const float*)d_in[7];
    const float* b_dt  = (const float*)d_in[8];
    const float* alog  = (const float*)d_in[9];
    const float* Ds    = (const float*)d_in[10];
    const float* olnw  = (const float*)d_in[11];
    const float* olnb  = (const float*)d_in[12];
    const float* w_out = (const float*)d_in[13];
    float* out = (float*)d_out;

    // workspace layout (256B-aligned, ~44 MB total)
    char* ws = (char*)d_ws;
    _Float16* w1h  = (_Float16*)(ws + 0);          // 512*128 f16
    _Float16* w2h  = (_Float16*)(ws + 131072);     // 4*40*256 f16
    _Float16* w3h  = (_Float16*)(ws + 212992);     // 128*256 f16
    _Float16* xn16 = (_Float16*)(ws + 278528);     // 9216*128 f16
    float*    x1   = (float*)   (ws + 2637824);    // (B,DI,L) f32
    float*    z    = (float*)   (ws + 12075008);   // (B,L,DI) f32
    _Float16* xc16 = (_Float16*)(ws + 21512192);   // (B,L,DI) f16
    float*    xdbl = (float*)   (ws + 26230784);   // (B,K,40,L) f32
    float*    ymg  = (float*)   (ws + 32129024);   // (B,L,DI) f32
    _Float16* yln  = (_Float16*)(ws + 41566208);   // (B,L,DI) f16

    cvt16<<<(512 * 128 + 255) / 256, 256, 0, stream>>>(w_in, w1h, 512 * 128);
    cvt16<<<(cK * 40 * cDI + 255) / 256, 256, 0, stream>>>(w_xp, w2h, cK * 40 * cDI);
    cvt16<<<(cC * cDI + 255) / 256, 256, 0, stream>>>(w_out, w3h, cC * cDI);

    ln_in<<<(cNPIX + 255) / 256, 256, 0, stream>>>(x, lnw, lnb, xn16);
    gemm_inproj<<<dim3(cNPIX / 128, 512 / 16), 256, 0, stream>>>(xn16, w1h, x1, z);
    dwconv<<<dim3(cL / 256, cB * cDI), 256, 0, stream>>>(x1, cw, cb, xc16);
    gemm_xproj<<<dim3(cL / 128, 3, cB * cK), 256, 0, stream>>>(xc16, w2h, xdbl);

    hipMemsetAsync(ymg, 0, (size_t)cNPIX * cDI * sizeof(float), stream);
    sel_scan<<<cB * cK * (cDI / 32), 32, 0, stream>>>(xdbl, xc16, w_dt, b_dt,
                                                      alog, Ds, ymg);

    ln_out<<<(cNPIX + 255) / 256, 256, 0, stream>>>(ymg, olnw, olnb, z, yln);
    gemm_outproj<<<dim3(cNPIX / 128, cC / 16), 256, 0, stream>>>(yln, w3h, x, out);
}
